// GPT3_56934086476265
// MI455X (gfx1250) — compile-verified, matrix operand order checked by reading, not run on hardware
//
#include <hip/hip_runtime.h>
#include <math.h>

// ---------------------------------------------------------------------------
// MI455X (gfx1250) transformer pipeline.
//  * All GEMM operands pre-converted to bf16; weights pre-transposed so both
//    A and B are K-major [row][k] -> LDS staging is a pure 16B copy.
//  * GEMM: 128x128 block tile (256 thr / 8 waves), 32x64 per wave,
//    v_wmma_f32_16x16x32_bf16, 8 WMMAs per wave per K=32 step.
//  * Double-buffered LDS with GLOBAL_LOAD_ASYNC_TO_LDS_B128 + S_WAIT_ASYNCCNT
//    (builtins probe-confirmed present; signature takes AS(1)/AS(3) v4i*).
// ---------------------------------------------------------------------------

typedef __attribute__((ext_vector_type(16))) __bf16 v16bf;
typedef __attribute__((ext_vector_type(8)))  __bf16 v8bf;
typedef __attribute__((ext_vector_type(4)))  __bf16 v4bf;
typedef __attribute__((ext_vector_type(8)))  float  v8f;
typedef int v4i_vs __attribute__((vector_size(4 * sizeof(int))));

#if defined(__has_builtin)
#if __has_builtin(__builtin_amdgcn_global_load_async_to_lds_b128) && \
    __has_builtin(__builtin_amdgcn_s_wait_asynccnt)
#define HAVE_ASYNC 1
#endif
#endif
#ifndef HAVE_ASYNC
#define HAVE_ASYNC 0
#endif

#define TILE_M 128
#define TILE_N 128
#define TILE_K 32

__device__ __forceinline__ void wait_async_lds() {
#if HAVE_ASYNC
    __builtin_amdgcn_s_wait_asynccnt(0);
#endif
}

// Copy a 128x32 bf16 tile (K-major rows) global -> LDS. 512 16B chunks,
// 2 per thread with 256 threads. Addresses are 16B aligned (all lds/ldb
// used are multiples of 8 elements; k0 multiple of 32).
__device__ __forceinline__ void stage128x32(const __bf16* __restrict__ g, int ld,
                                            int r0, int k0, __bf16* l, int tid)
{
#pragma unroll
    for (int it = 0; it < 2; ++it) {
        int c = tid + it * 256;
        int r = c >> 2, off = (c & 3) * 8;
        const __bf16* src = g + (long)(r0 + r) * ld + (k0 + off);
        __bf16* dst = l + r * TILE_K + off;
#if HAVE_ASYNC
        __builtin_amdgcn_global_load_async_to_lds_b128(
            (__attribute__((address_space(1))) v4i_vs*)src,
            (__attribute__((address_space(3))) v4i_vs*)dst, 0, 0);
#else
        *(v8bf*)dst = *(const v8bf*)src;
#endif
    }
}

__device__ __forceinline__ v16bf load_frag(const __bf16* p) {
    // CDNA5 16-bit A/B lane layout: elements 0..7 -> K = half*8+e,
    // elements 8..15 -> K = 16+half*8+(e-8); two 16B LDS loads.
    v8bf lo = *(const v8bf*)(p);
    v8bf hi = *(const v8bf*)(p + 16);
    return __builtin_shufflevector(lo, hi, 0, 1, 2, 3, 4, 5, 6, 7,
                                   8, 9, 10, 11, 12, 13, 14, 15);
}

// C = A[M][K] * B^T (B stored [N][K]) + bias + res.
// outMode: 0 = bf16 C[M][N]; 1 = f32 C[M][N]; 2 = bf16 transposed C[N][M]
// (ldc is the row stride of whichever layout is written; res uses ldc, only
// valid for non-transposed output).
__global__ __launch_bounds__(256)
void gemm_bt_wmma(const __bf16* __restrict__ A, int lda,
                  const __bf16* __restrict__ B, int ldb,
                  const float* __restrict__ bias,
                  const __bf16* __restrict__ res,
                  void* __restrict__ Cout, int ldc, int K, int outMode)
{
    __shared__ __align__(16) __bf16 sa[2][TILE_M * TILE_K];
    __shared__ __align__(16) __bf16 sb[2][TILE_N * TILE_K];

    const int tid  = threadIdx.x;
    const int lane = tid & 31;
    const int wave = tid >> 5;
    const int half = lane >> 4;
    const int lrow = lane & 15;

    const int m0 = blockIdx.y * TILE_M;
    const int n0 = blockIdx.x * TILE_N;
    const int wm = (wave >> 1) * 32;  // 4 wave rows
    const int wn = (wave & 1) * 64;   // 2 wave cols

    v8f acc[2][4] = {};

    stage128x32(A, lda, m0, 0, sa[0], tid);
    stage128x32(B, ldb, n0, 0, sb[0], tid);
    wait_async_lds();
    __syncthreads();

    int cur = 0;
    for (int k0 = 0; k0 < K; k0 += TILE_K) {
        const int nxt = cur ^ 1;
        if (k0 + TILE_K < K) {
            stage128x32(A, lda, m0, k0 + TILE_K, sa[nxt], tid);
            stage128x32(B, ldb, n0, k0 + TILE_K, sb[nxt], tid);
        }

        v16bf af[2], bfr[4];
#pragma unroll
        for (int t = 0; t < 2; ++t)
            af[t] = load_frag(&sa[cur][(wm + t * 16 + lrow) * TILE_K + half * 8]);
#pragma unroll
        for (int t = 0; t < 4; ++t)
            bfr[t] = load_frag(&sb[cur][(wn + t * 16 + lrow) * TILE_K + half * 8]);

#pragma unroll
        for (int tm = 0; tm < 2; ++tm)
#pragma unroll
            for (int tn = 0; tn < 4; ++tn)
                acc[tm][tn] = __builtin_amdgcn_wmma_f32_16x16x32_bf16(
                    false, af[tm], false, bfr[tn],
                    (short)0, acc[tm][tn], false, false);

        wait_async_lds();   // next tile fully in LDS (overlapped with WMMAs)
        __syncthreads();
        cur = nxt;
    }

    // C/D layout: lane l -> N = l%16; VGPR r -> M = (l/16)*8 + r.
#pragma unroll
    for (int tm = 0; tm < 2; ++tm)
#pragma unroll
        for (int tn = 0; tn < 4; ++tn)
#pragma unroll
            for (int r = 0; r < 8; ++r) {
                int mm = m0 + wm + tm * 16 + half * 8 + r;
                int nn = n0 + wn + tn * 16 + lrow;
                float v = acc[tm][tn][r];
                if (bias) v += bias[nn];
                if (res)  v += (float)res[(long)mm * ldc + nn];
                if (outMode == 1)
                    ((float*)Cout)[(long)mm * ldc + nn] = v;
                else if (outMode == 0)
                    ((__bf16*)Cout)[(long)mm * ldc + nn] = (__bf16)v;
                else
                    ((__bf16*)Cout)[(long)nn * ldc + mm] = (__bf16)v;
            }
}

// ---------------------------------------------------------------------------
// Conversion kernels (one-time prologue).
// ---------------------------------------------------------------------------

__global__ __launch_bounds__(256)
void cvt_f32_bf16(const float* __restrict__ in, __bf16* __restrict__ out, long n)
{
    long i = ((long)blockIdx.x * 256 + threadIdx.x) * 4;
    if (i < n) {
        const float4 v = *(const float4*)(in + i);
        v4bf o;
        o[0] = (__bf16)v.x; o[1] = (__bf16)v.y; o[2] = (__bf16)v.z; o[3] = (__bf16)v.w;
        *(v4bf*)(out + i) = o;
    }
}

// out[c][r] = (bf16) in[r][c]; rows, cols multiples of 32.
__global__ __launch_bounds__(256)
void cvt_transpose_bf16(const float* __restrict__ in, __bf16* __restrict__ out,
                        int rows, int cols)
{
    __shared__ float tile[32][33];
    const int tx = threadIdx.x & 31;
    const int ty = threadIdx.x >> 5;
    const int by = blockIdx.y * 32;  // row base
    const int bx = blockIdx.x * 32;  // col base
#pragma unroll
    for (int j = 0; j < 4; ++j)
        tile[ty + j * 8][tx] = in[(long)(by + ty + j * 8) * cols + (bx + tx)];
    __syncthreads();
#pragma unroll
    for (int j = 0; j < 4; ++j)
        out[(long)(bx + ty + j * 8) * rows + (by + tx)] = (__bf16)tile[tx][ty + j * 8];
}

// ---------------------------------------------------------------------------
// Row kernels (bf16 activations, f32 math).
// ---------------------------------------------------------------------------

__global__ __launch_bounds__(256)
void softmax_row_1024(__bf16* __restrict__ S)
{
    __shared__ float red[256];
    __bf16* row = S + (long)blockIdx.x * 1024;
    const int tid = threadIdx.x;
    float v[4];
    float m = -3.402823466e38f;
#pragma unroll
    for (int j = 0; j < 4; ++j) { v[j] = (float)row[tid + 256 * j]; m = fmaxf(m, v[j]); }
    red[tid] = m; __syncthreads();
    for (int s = 128; s > 0; s >>= 1) {
        if (tid < s) red[tid] = fmaxf(red[tid], red[tid + s]);
        __syncthreads();
    }
    m = red[0]; __syncthreads();
    float sum = 0.f;
#pragma unroll
    for (int j = 0; j < 4; ++j) { v[j] = __expf(v[j] - m); sum += v[j]; }
    red[tid] = sum; __syncthreads();
    for (int s = 128; s > 0; s >>= 1) {
        if (tid < s) red[tid] += red[tid + s];
        __syncthreads();
    }
    const float inv = 1.0f / red[0];
#pragma unroll
    for (int j = 0; j < 4; ++j) row[tid + 256 * j] = (__bf16)(v[j] * inv);
}

__global__ __launch_bounds__(256)
void l2norm_row_768(const __bf16* __restrict__ in, __bf16* __restrict__ out)
{
    __shared__ float red[256];
    const __bf16* r = in + (long)blockIdx.x * 768;
    __bf16* o = out + (long)blockIdx.x * 768;
    const int tid = threadIdx.x;
    float v[3];
    float s = 0.f;
#pragma unroll
    for (int j = 0; j < 3; ++j) { v[j] = (float)r[tid + 256 * j]; s += v[j] * v[j]; }
    red[tid] = s; __syncthreads();
    for (int st = 128; st > 0; st >>= 1) {
        if (tid < st) red[tid] += red[tid + st];
        __syncthreads();
    }
    const float inv = 1.0f / fmaxf(sqrtf(red[0]), 1e-12f);
#pragma unroll
    for (int j = 0; j < 3; ++j) o[tid + 256 * j] = (__bf16)(v[j] * inv);
}

// Top-10 along the sequence dim per vocab column, sorted descending.
__global__ __launch_bounds__(128)
void topk10_col(const float* __restrict__ logits, float* __restrict__ out)
{
    const int col = blockIdx.x * 128 + threadIdx.x;
    float best[10];
#pragma unroll
    for (int j = 0; j < 10; ++j) best[j] = -3.402823466e38f;
    for (int n = 0; n < 1024; ++n) {
        if (n + 8 < 1024)
            __builtin_prefetch(logits + (long)(n + 8) * 32000 + col, 0, 0);
        float v = logits[(long)n * 32000 + col];
        if (v > best[9]) {
#pragma unroll
            for (int j = 0; j < 10; ++j) {
                if (v > best[j]) { float t = best[j]; best[j] = v; v = t; }
            }
        }
    }
#pragma unroll
    for (int j = 0; j < 10; ++j) out[(long)j * 32000 + col] = best[j];
}

// ---------------------------------------------------------------------------

extern "C" void kernel_launch(void* const* d_in, const int* in_sizes, int n_in,
                              void* d_out, int out_size, void* d_ws, size_t ws_size,
                              hipStream_t stream)
{
    (void)in_sizes; (void)n_in; (void)out_size; (void)ws_size;

    const float* x    = (const float*)d_in[0];   // [1024,32000]
    const float* pe   = (const float*)d_in[1];   // [1024,768]
    const float* Wemb = (const float*)d_in[2];   // [32000,768]
    const float* bemb = (const float*)d_in[3];
    const float* Wq   = (const float*)d_in[4];   // [768,128]
    const float* bq   = (const float*)d_in[5];
    const float* Wk   = (const float*)d_in[6];
    const float* bk   = (const float*)d_in[7];
    const float* Wv   = (const float*)d_in[8];
    const float* bv   = (const float*)d_in[9];
    const float* Wo   = (const float*)d_in[10];  // [128,768]
    const float* bo   = (const float*)d_in[11];
    const float* W1   = (const float*)d_in[12];  // [768,768]
    const float* b1   = (const float*)d_in[13];
    const float* Wout = (const float*)d_in[14];  // [768,32000]
    const float* bout = (const float*)d_in[15];
    // d_in[16] = k (10); output shape fixed by out_size = 10*32000.

    char* p = (char*)d_ws;
    auto carve_bf16 = [&](long n) { __bf16* q = (__bf16*)p; p += n * 2; return q; };
    auto carve_f32  = [&](long n) { float*  q = (float*)p;  p += n * 4; return q; };

    __bf16* xb     = carve_bf16((long)1024 * 32000);
    __bf16* peb    = carve_bf16(1024 * 768);
    __bf16* Wembt  = carve_bf16((long)32000 * 768);   // [768][32000]
    __bf16* Woutt  = carve_bf16((long)32000 * 768);   // [32000][768]
    __bf16* Wqt    = carve_bf16(768 * 128);           // [128][768]
    __bf16* Wkt    = carve_bf16(768 * 128);
    __bf16* Wvt    = carve_bf16(768 * 128);
    __bf16* Wot    = carve_bf16(768 * 128);           // [768][128]
    __bf16* W1t    = carve_bf16(768 * 768);           // [768][768]
    __bf16* h      = carve_bf16(1024 * 768);
    __bf16* t1     = carve_bf16(1024 * 768);
    __bf16* n1     = carve_bf16(1024 * 768);
    __bf16* r2     = carve_bf16(1024 * 768);
    __bf16* t2     = carve_bf16(1024 * 768);
    __bf16* Q      = carve_bf16(1024 * 128);
    __bf16* Kb     = carve_bf16(1024 * 128);
    __bf16* Vt     = carve_bf16(1024 * 128);          // [128][1024]
    __bf16* A1     = carve_bf16(1024 * 128);
    __bf16* S      = carve_bf16((long)1024 * 1024);
    float*  logits = carve_f32((long)1024 * 32000);

    // ---- one-time bf16 conversion / weight transposition -------------------
    {
        long nx = (long)1024 * 32000;
        cvt_f32_bf16<<<dim3((nx / 4 + 255) / 256), 256, 0, stream>>>(x, xb, nx);
        long npe = 1024 * 768;
        cvt_f32_bf16<<<dim3((npe / 4 + 255) / 256), 256, 0, stream>>>(pe, peb, npe);
        cvt_transpose_bf16<<<dim3(768 / 32, 32000 / 32), 256, 0, stream>>>(Wemb, Wembt, 32000, 768);
        cvt_transpose_bf16<<<dim3(32000 / 32, 768 / 32), 256, 0, stream>>>(Wout, Woutt, 768, 32000);
        cvt_transpose_bf16<<<dim3(128 / 32, 768 / 32), 256, 0, stream>>>(Wq, Wqt, 768, 128);
        cvt_transpose_bf16<<<dim3(128 / 32, 768 / 32), 256, 0, stream>>>(Wk, Wkt, 768, 128);
        cvt_transpose_bf16<<<dim3(128 / 32, 768 / 32), 256, 0, stream>>>(Wv, Wvt, 768, 128);
        cvt_transpose_bf16<<<dim3(768 / 32, 128 / 32), 256, 0, stream>>>(Wo, Wot, 128, 768);
        cvt_transpose_bf16<<<dim3(768 / 32, 768 / 32), 256, 0, stream>>>(W1, W1t, 768, 768);
    }

    auto gemm = [&](const __bf16* A, int lda, const __bf16* B, int ldb,
                    const float* bias, const __bf16* res, void* C, int ldc,
                    int M, int N, int K, int outMode) {
        dim3 grid(N / TILE_N, M / TILE_M);
        gemm_bt_wmma<<<grid, dim3(256), 0, stream>>>(A, lda, B, ldb, bias, res,
                                                     C, ldc, K, outMode);
    };

    // h = x @ Wemb + bemb + pe
    gemm(xb, 32000, Wembt, 32000, bemb, peb, h, 768, 1024, 768, 32000, 0);

    for (int blk = 0; blk < 96; ++blk) {
        gemm(h, 768, Wqt, 768, bq, nullptr, Q,  128, 1024, 128, 768, 0);
        gemm(h, 768, Wkt, 768, bk, nullptr, Kb, 128, 1024, 128, 768, 0);
        gemm(h, 768, Wvt, 768, bv, nullptr, Vt, 1024, 1024, 128, 768, 2); // V^T
        // S = Q @ K^T  (K already [n][k] layout)
        gemm(Q, 128, Kb, 128, nullptr, nullptr, S, 1024, 1024, 1024, 128, 0);
        softmax_row_1024<<<1024, 256, 0, stream>>>(S);
        // A1 = S @ V   (B = V^T, [128][1024])
        gemm(S, 1024, Vt, 1024, nullptr, nullptr, A1, 128, 1024, 128, 1024, 0);
        // t1 = A1 @ Wo + bo + h
        gemm(A1, 128, Wot, 128, bo, h, t1, 768, 1024, 768, 128, 0);
        l2norm_row_768<<<1024, 256, 0, stream>>>(t1, n1);
        // r2 = n1 + n1 @ W1 + b1
        gemm(n1, 768, W1t, 768, b1, n1, r2, 768, 1024, 768, 768, 0);
        // t2 = r2 @ W1 + b1 ; h = l2norm(t2)
        gemm(r2, 768, W1t, 768, b1, nullptr, t2, 768, 1024, 768, 768, 0);
        l2norm_row_768<<<1024, 256, 0, stream>>>(t2, h);
    }

    // logits = h @ Wout + bout (f32 for top-k)
    gemm(h, 768, Woutt, 768, bout, nullptr, logits, 32000, 1024, 32000, 768, 1);
    topk10_col<<<250, 128, 0, stream>>>(logits, (float*)d_out);
}
